// GCNJointRepresentation_84129819394297
// MI455X (gfx1250) — matrix operand order repaired and checked by reading, not compiled, not run on hardware
//
#include <hip/hip_runtime.h>
#include <hip/hip_bf16.h>

typedef __attribute__((ext_vector_type(16))) __bf16 v16bf;
typedef __attribute__((ext_vector_type(8)))  float  v8f;

#define ZD        64
#define AD        768
#define KTOT      832            // ZD + AD
#define HID       128
#define NC        5
#define KT_TILES  26             // KTOT / 32
#define KT_HALF   13             // staged half of the K range
#define BE        64             // edges per block (4 M-tiles of 16)
#define NTILES    8              // HID / 16
#define THREADS   128            // 4 waves: 2(M) x 2(N)

// ---- optional gfx1250 async global->LDS copy (probe via __has_builtin) ----
#if defined(__has_builtin)
#  if __has_builtin(__builtin_amdgcn_global_load_async_to_lds_b128)
#    define HAVE_ASYNC_COPY 1
#  endif
#endif

typedef __attribute__((__vector_size__(4 * sizeof(int)))) int i4v;
typedef __attribute__((address_space(1))) i4v* as1_i4v_p;
typedef __attribute__((address_space(3))) i4v* as3_i4v_p;

__device__ __forceinline__ void async_copy_b128(const void* g, void* l) {
#if defined(HAVE_ASYNC_COPY)
  __builtin_amdgcn_global_load_async_to_lds_b128(
      (as1_i4v_p)(void*)g, (as3_i4v_p)l, 0, 0);
#else
  *(uint4*)l = *(const uint4*)g;
#endif
}

__device__ __forceinline__ void wait_async0() {
#if defined(HAVE_ASYNC_COPY)
#  if __has_builtin(__builtin_amdgcn_s_wait_asynccnt)
  __builtin_amdgcn_s_wait_asynccnt(0);
#  else
  asm volatile("s_wait_asynccnt 0" ::: "memory");
#  endif
#endif
}

// d_ws layout: W1 repacked to bf16 WMMA B-fragments.
//   element index = ((kt*8 + nt)*32 + lane)*16 + h
//   k = kt*32 + 16*(lane>>4) + h ;  n = nt*16 + (lane&15)
#define W1F_HALVES (KT_TILES * NTILES * 32 * 16)   // 106496

__global__ __launch_bounds__(256) void repack_w1(const float* __restrict__ W1,
                                                 __bf16* __restrict__ w1f) {
  int idx = blockIdx.x * 256 + threadIdx.x;
  if (idx >= W1F_HALVES) return;
  int h    = idx & 15;
  int lane = (idx >> 4) & 31;
  int nt   = (idx >> 9) & 7;
  int kt   = idx >> 12;
  int k = kt * 32 + ((lane >> 4) << 4) + h;
  int n = nt * 16 + (lane & 15);
  w1f[idx] = (__bf16)W1[k * HID + n];
}

__global__ __launch_bounds__(THREADS) void fused_kernel(
    const float* __restrict__ z,
    const long long* __restrict__ eidx,
    const float* __restrict__ ea,
    const __bf16* __restrict__ w1f,
    const float* __restrict__ b1,
    const float* __restrict__ W2,
    const float* __restrict__ b2,
    float* __restrict__ out,
    int E) {
  // A-fragments (bf16) for one K-half: 13 ktiles * 4 mtiles * 32 lanes * 16 halves
  __shared__ __bf16 stage[KT_HALF * 4 * 32 * 16];    // 53248 B
  __shared__ float  w2s[HID * NC];
  __shared__ float  b1s[HID];
  __shared__ float  b2s[NC];

  const int tid   = threadIdx.x;
  const int lane  = tid & 31;
  const int w     = tid >> 5;
  const int wm    = w >> 1;          // 0..1 -> M subtile (32 edges)
  const int wn    = w & 1;           // 0..1 -> N subtile (64 cols)
  const int khalf = lane >> 4;
  const int mlo   = lane & 15;
  const int e0    = blockIdx.x * BE;

  // preload small weights into LDS (async path if the toolchain exposes it)
  for (int i = tid; i < (HID * NC) / 4; i += THREADS)   // 160 x 16B
    async_copy_b128(W2 + i * 4, &w2s[i * 4]);
  for (int i = tid; i < HID / 4; i += THREADS)          // 32 x 16B
    async_copy_b128(b1 + i * 4, &b1s[i * 4]);
  if (tid < NC) b2s[tid] = b2[tid];
  wait_async0();

  v8f zero = {0.f, 0.f, 0.f, 0.f, 0.f, 0.f, 0.f, 0.f};
  v8f acc[2][4];
  #pragma unroll
  for (int mi = 0; mi < 2; ++mi)
    #pragma unroll
    for (int ni = 0; ni < 4; ++ni) acc[mi][ni] = zero;

  for (int s = 0; s < 2; ++s) {
    __syncthreads();   // previous compute phase (or weight preload) done
    // ---- fill A-fragment stage for K tiles [s*13, s*13+13) ----
    // one group = 8 consecutive K values for one (ktl, mt, lane, blk)
    for (int g = tid; g < KT_HALF * 4 * 32 * 2; g += THREADS) {
      int blk   = g & 1;
      int glane = (g >> 1) & 31;
      int mt    = (g >> 6) & 3;
      int ktl   = g >> 8;
      int kt    = s * KT_HALF + ktl;
      int kb    = kt * 32 + 16 * (glane >> 4) + 8 * blk;   // first K of the 8
      int r     = mt * 16 + (glane & 15);                  // edge row in block
      long long e = (long long)e0 + r;
      if (e >= E) e = E - 1;
      float v[8];
      if (kb < ZD) {
        long long srcn = eidx[e];
        long long dstn = eidx[(long long)E + e];
        const float4* zs = (const float4*)(z + srcn * ZD + kb);
        const float4* zd = (const float4*)(z + dstn * ZD + kb);
        float4 a0 = zs[0], a1 = zs[1], d0 = zd[0], d1 = zd[1];
        v[0]=a0.x*d0.x; v[1]=a0.y*d0.y; v[2]=a0.z*d0.z; v[3]=a0.w*d0.w;
        v[4]=a1.x*d1.x; v[5]=a1.y*d1.y; v[6]=a1.z*d1.z; v[7]=a1.w*d1.w;
      } else {
        const float4* ap = (const float4*)(ea + e * (long long)AD + (kb - ZD));
        float4 a0 = ap[0], a1 = ap[1];
        v[0]=a0.x; v[1]=a0.y; v[2]=a0.z; v[3]=a0.w;
        v[4]=a1.x; v[5]=a1.y; v[6]=a1.z; v[7]=a1.w;
      }
      union { __bf16 h[8]; uint4 q; } pk;
      #pragma unroll
      for (int i = 0; i < 8; ++i) pk.h[i] = (__bf16)v[i];
      int off = (((ktl * 4 + mt) * 32 + glane) << 4) + (blk << 3);
      *(uint4*)&stage[off] = pk.q;
    }
    __syncthreads();

    if (s == 0) {
      // prefetch stage-1 edge_attr lines (K 416..831 -> ea cols 352..767)
      // 13 chunks of 128B per edge; fire-and-forget global_prefetch_b8
      for (int p = tid; p < BE * 13; p += THREADS) {
        int eLoc  = p / 13;
        int chunk = p - eLoc * 13;
        long long e = (long long)e0 + eLoc;
        if (e >= E) e = E - 1;
        __builtin_prefetch(ea + e * (long long)AD + 352 + chunk * 32, 0, 1);
      }
    }

    // ---- WMMA over this K half: each wave 32M x 64N = 8 WMMAs / k-tile ----
    for (int ktl = 0; ktl < KT_HALF; ++ktl) {
      int kt = s * KT_HALF + ktl;
      v16bf afr[2], bfr[4];
      #pragma unroll
      for (int mi = 0; mi < 2; ++mi) {
        int mt = wm * 2 + mi;
        afr[mi] = *(const v16bf*)&stage[(((ktl * 4 + mt) * 32 + lane) << 4)];
      }
      #pragma unroll
      for (int ni = 0; ni < 4; ++ni) {
        int nt = wn * 4 + ni;
        bfr[ni] = *(const v16bf*)&w1f[(((kt * 8 + nt) * 32 + lane) << 4)];
      }
      #pragma unroll
      for (int mi = 0; mi < 2; ++mi)
        #pragma unroll
        for (int ni = 0; ni < 4; ++ni)
          acc[mi][ni] = __builtin_amdgcn_wmma_f32_16x16x32_bf16(
              false, afr[mi], false, bfr[ni], (short)0, acc[mi][ni],
              false, false);
    }
  }

  // ---- epilogue: h = relu(acc + b1) -> LDS (stride 133 floats, conflict-free)
  __syncthreads();
  float* hBuf = (float*)stage;        // 64 * 133 * 4 B = 34048 B, fits in stage
  #pragma unroll
  for (int mi = 0; mi < 2; ++mi) {
    #pragma unroll
    for (int ni = 0; ni < 4; ++ni) {
      int n = (wn * 4 + ni) * 16 + mlo;
      float bias = b1s[n];
      #pragma unroll
      for (int j = 0; j < 8; ++j) {
        int rr = (wm * 2 + mi) * 16 + j + 8 * khalf;   // edge row in block
        float hv = acc[mi][ni][j] + bias;
        hBuf[rr * 133 + n] = fmaxf(hv, 0.f);
      }
    }
  }
  __syncthreads();

  // ---- GEMM2 (128x5) + softmax, one thread per edge ----
  if (tid < BE) {
    long long e = (long long)e0 + tid;
    if (e < E) {
      float lg[NC];
      #pragma unroll
      for (int c = 0; c < NC; ++c) lg[c] = b2s[c];
      for (int k = 0; k < HID; ++k) {
        float hv = hBuf[tid * 133 + k];
        #pragma unroll
        for (int c = 0; c < NC; ++c) lg[c] += hv * w2s[k * NC + c];
      }
      float m = lg[0];
      #pragma unroll
      for (int c = 1; c < NC; ++c) m = fmaxf(m, lg[c]);
      float sum = 0.f;
      #pragma unroll
      for (int c = 0; c < NC; ++c) { lg[c] = __expf(lg[c] - m); sum += lg[c]; }
      float inv = 1.f / sum;
      #pragma unroll
      for (int c = 0; c < NC; ++c) out[e * NC + c] = lg[c] * inv;
    }
  }
}

extern "C" void kernel_launch(void* const* d_in, const int* in_sizes, int n_in,
                              void* d_out, int out_size, void* d_ws, size_t ws_size,
                              hipStream_t stream) {
  const float*     z    = (const float*)d_in[0];
  const long long* eidx = (const long long*)d_in[1];
  const float*     ea   = (const float*)d_in[2];
  const float*     W1   = (const float*)d_in[3];
  const float*     b1   = (const float*)d_in[4];
  const float*     W2   = (const float*)d_in[5];
  const float*     b2   = (const float*)d_in[6];
  float*           out  = (float*)d_out;

  int E = in_sizes[2] / AD;       // number of edges from edge_attr
  __bf16* w1f = (__bf16*)d_ws;    // 212992 bytes of workspace

  repack_w1<<<W1F_HALVES / 256, 256, 0, stream>>>(W1, w1f);
  int blocks = (E + BE - 1) / BE;
  fused_kernel<<<blocks, THREADS, 0, stream>>>(z, eidx, ea, w1f, b1, W2, b2, out, E);
}
